// PlanningModule_70987219468591
// MI455X (gfx1250) — compile-verified, matrix operand order checked by reading, not run on hardware
//
#include <hip/hip_runtime.h>
#include <cstdint>
#include <cstddef>

// Problem constants (match reference setup_inputs)
#define BQ 16
#define SQ 2048
#define HQ 2048
#define PQ 2048
#define STEPSQ 5
#define EPSQ 1e-5f

typedef __attribute__((ext_vector_type(8))) int v8i;

// ---------------------------------------------------------------------------
// Utility kernels
// ---------------------------------------------------------------------------

__global__ void init_slots_kernel(float* sums) {
  if (threadIdx.x < 16) sums[threadIdx.x] = 0.f;
}

__global__ __launch_bounds__(256)
void abssum_kernel(const float* __restrict__ w, long long n, float* __restrict__ slot) {
  __shared__ float red[256];
  float s = 0.f;
  for (long long i = (long long)blockIdx.x * 256 + threadIdx.x; i < n;
       i += (long long)gridDim.x * 256)
    s += fabsf(w[i]);
  red[threadIdx.x] = s;
  __syncthreads();
  for (int st = 128; st > 0; st >>= 1) {
    if (threadIdx.x < st) red[threadIdx.x] += red[threadIdx.x + st];
    __syncthreads();
  }
  if (threadIdx.x == 0) atomicAdd(slot, red[0]);
}

// BitNet ternary weight quantization: wq = clip(round(w/gamma),-1,1), gamma=mean|w|
__global__ __launch_bounds__(256)
void quantize_weight_kernel(const float* __restrict__ w, long long n,
                            const float* __restrict__ sumSlot,
                            float* __restrict__ gammaSlot,
                            int8_t* __restrict__ wq) {
  float gamma = fmaxf(sumSlot[0] / (float)n, EPSQ);
  long long i = (long long)blockIdx.x * 256 + threadIdx.x;
  if (i < n) {
    float q = rintf(w[i] / gamma);
    q = fminf(fmaxf(q, -1.f), 1.f);
    wq[i] = (int8_t)q;
  }
  if (blockIdx.x == 0 && threadIdx.x == 0) gammaSlot[0] = gamma;
}

// Per-row int8 absmax quantization: s = 127/max(|x|,eps); store xq and 1/s
__global__ __launch_bounds__(256)
void quantize_rows_kernel(const float* __restrict__ x, int cols,
                          int8_t* __restrict__ xq, float* __restrict__ rscale) {
  const int row = blockIdx.x;
  const float* xr = x + (size_t)row * cols;
  __shared__ float red[256];
  float m = 0.f;
  for (int c = threadIdx.x; c < cols; c += 256) m = fmaxf(m, fabsf(xr[c]));
  red[threadIdx.x] = m;
  __syncthreads();
  for (int st = 128; st > 0; st >>= 1) {
    if (threadIdx.x < st) red[threadIdx.x] = fmaxf(red[threadIdx.x], red[threadIdx.x + st]);
    __syncthreads();
  }
  float mx = fmaxf(red[0], EPSQ);
  float s = 127.f / mx;
  int8_t* xqr = xq + (size_t)row * cols;
  for (int c = threadIdx.x; c < cols; c += 256) {
    float q = rintf(xr[c] * s);
    q = fminf(fmaxf(q, -128.f), 127.f);
    xqr[c] = (int8_t)q;
  }
  if (threadIdx.x == 0) rscale[row] = mx / 127.f;  // = 1/s (dequant factor)
}

// ---------------------------------------------------------------------------
// Core GEMM: int8 x int8 -> int32 via V_WMMA_I32_16X16X64_IU8
// out[m,n] = (sum_k Xq[m,k]*Wq[n,k]) * gamma_w * rowscale[m] + bias[n]  (opt tanh)
//
// Block tile 64(M) x 128(N): 8 waves arranged 2(M) x 4(N); each wave owns a
// 32x32 tile = 2x2 WMMA accumulators. The 4 independent back-to-back WMMAs per
// K-chunk fill each other's RAW hazard slots (WMMA_IU8 -> VALU needs 8 coexec
// slots) instead of v_nop padding, and amortize LDS fragment traffic and the
// workgroup barrier 4x relative to a single-tile wave.
// LDS double-buffered (24KB), filled with CDNA5 async global->LDS loads.
// ---------------------------------------------------------------------------

#define GT_M 64
#define GT_N 128
#define GT_K 64

__global__ __launch_bounds__(256)
void gemm_i8_wmma_kernel(const int8_t* __restrict__ Xq, const float* __restrict__ xscale,
                         const int8_t* __restrict__ Wq, const float* __restrict__ gammaSlot,
                         const float* __restrict__ bias,
                         int M, int N, int K, int act,
                         float* __restrict__ outF32, _Float16* __restrict__ outF16) {
  __shared__ alignas(32) int8_t lA[2][GT_M][GT_K];   // 2 x 4KB
  __shared__ alignas(32) int8_t lB[2][GT_N][GT_K];   // 2 x 8KB

  const int t = threadIdx.x;
  const int wave = t >> 5;
  const int lane = t & 31;
  const int wm = wave >> 2;        // 0..1 -> M sub-tile (32 rows)
  const int wn = wave & 3;         // 0..3 -> N sub-tile (32 cols)
  const int rowBase = blockIdx.y * GT_M;
  const int colBase = blockIdx.x * GT_N;
  const int nk = K / GT_K;

  auto load_tiles = [&](int kc, int buf) {
    // A tile: 64 rows x 64B = 4096B -> 16B (b128) per thread, async to LDS
    {
      int aoff = t * 16;
      int r = aoff >> 6, c = aoff & 63;
      int gr = rowBase + r;
      if (gr >= M) gr = M - 1;                 // clamp (results masked in epilogue)
      const int8_t* src = Xq + (size_t)gr * K + (size_t)kc * GT_K + c;
      unsigned lds = (unsigned)(uintptr_t)&lA[buf][r][c];
      unsigned long long ga = (unsigned long long)(uintptr_t)src;
      asm volatile("global_load_async_to_lds_b128 %0, %1, off"
                   :: "v"(lds), "v"(ga) : "memory");
    }
    // B tile (weight rows == W^T columns): 128 rows x 64B = 8192B -> 2x b128
    {
      int boff = t * 32;
      int r = boff >> 6, c = boff & 63;
      const int8_t* src = Wq + (size_t)(colBase + r) * K + (size_t)kc * GT_K + c;
      unsigned lds = (unsigned)(uintptr_t)&lB[buf][r][c];
      unsigned long long ga = (unsigned long long)(uintptr_t)src;
      asm volatile("global_load_async_to_lds_b128 %0, %1, off"
                   :: "v"(lds), "v"(ga) : "memory");
      unsigned lds2 = lds + 16u;
      unsigned long long ga2 = ga + 16ull;
      asm volatile("global_load_async_to_lds_b128 %0, %1, off"
                   :: "v"(lds2), "v"(ga2) : "memory");
      if (kc + 2 < nk) __builtin_prefetch(src + 2 * GT_K, 0, 1);
    }
  };

  v8i acc00 = {0, 0, 0, 0, 0, 0, 0, 0};
  v8i acc01 = {0, 0, 0, 0, 0, 0, 0, 0};
  v8i acc10 = {0, 0, 0, 0, 0, 0, 0, 0};
  v8i acc11 = {0, 0, 0, 0, 0, 0, 0, 0};

  load_tiles(0, 0);
  asm volatile("s_wait_asynccnt 0" ::: "memory");
  __syncthreads();

  const int aRow0 = wm * 32 + (lane & 15);
  const int aRow1 = aRow0 + 16;
  const int bCol0 = wn * 32 + (lane & 15);
  const int bCol1 = bCol0 + 16;
  const int kOff = (lane >> 4) * 32;           // low/high lane half -> K halves

  for (int kc = 0; kc < nk; ++kc) {
    const int buf = kc & 1;
    if (kc + 1 < nk) load_tiles(kc + 1, buf ^ 1);

    v8i a0 = *(const v8i*)&lA[buf][aRow0][kOff];
    v8i a1 = *(const v8i*)&lA[buf][aRow1][kOff];
    v8i b0 = *(const v8i*)&lB[buf][bCol0][kOff];
    v8i b1 = *(const v8i*)&lB[buf][bCol1][kOff];

    acc00 = __builtin_amdgcn_wmma_i32_16x16x64_iu8(true, a0, true, b0, acc00, false, false);
    acc01 = __builtin_amdgcn_wmma_i32_16x16x64_iu8(true, a0, true, b1, acc01, false, false);
    acc10 = __builtin_amdgcn_wmma_i32_16x16x64_iu8(true, a1, true, b0, acc10, false, false);
    acc11 = __builtin_amdgcn_wmma_i32_16x16x64_iu8(true, a1, true, b1, acc11, false, false);

    asm volatile("s_wait_asynccnt 0" ::: "memory");
    __syncthreads();
  }

  // Epilogue: dequantize int32 accum -> float, optional bias / tanh
  const float gamma = gammaSlot[0];
  const int rhalf = (lane >> 4) * 8;           // C layout: lanes 16..31 hold M=r+8
  const int col0 = colBase + wn * 32 + (lane & 15);
  const int col1 = col0 + 16;
  const float bv0 = (bias != nullptr) ? bias[col0] : 0.f;
  const float bv1 = (bias != nullptr) ? bias[col1] : 0.f;

  v8i accs[2][2] = {{acc00, acc01}, {acc10, acc11}};
#pragma unroll
  for (int i = 0; i < 2; ++i) {
#pragma unroll
    for (int r = 0; r < 8; ++r) {
      int row = rowBase + wm * 32 + i * 16 + rhalf + r;
      if (row < M) {
        float xs = gamma * xscale[row];
        float v0 = (float)accs[i][0][r] * xs + bv0;
        float v1 = (float)accs[i][1][r] * xs + bv1;
        if (act == 1) { v0 = tanhf(v0); v1 = tanhf(v1); }
        if (outF16) {
          outF16[(size_t)row * N + col0] = (_Float16)v0;
          outF16[(size_t)row * N + col1] = (_Float16)v1;
        } else {
          outF32[(size_t)row * N + col0] = v0;
          outF32[(size_t)row * N + col1] = v1;
        }
      }
    }
  }
}

// ---------------------------------------------------------------------------
// Attention / GRU helper kernels
// ---------------------------------------------------------------------------

// scores[b,s] = scale * sum_p q[b,p]*K[b,s,p]; one wave per s
__global__ __launch_bounds__(256)
void scores_kernel(const float* __restrict__ q, const _Float16* __restrict__ K16,
                   float* __restrict__ scores, float scale) {
  const int b = blockIdx.y;
  const int wave = threadIdx.x >> 5;
  const int lane = threadIdx.x & 31;
  const int s = blockIdx.x * 8 + wave;
  const float* qb = q + (size_t)b * PQ;
  const _Float16* kr = K16 + ((size_t)b * SQ + s) * PQ;
  float acc = 0.f;
  for (int p = lane; p < PQ; p += 32) acc += qb[p] * (float)kr[p];
  for (int off = 16; off > 0; off >>= 1) acc += __shfl_down(acc, off, 32);
  if (lane == 0) scores[(size_t)b * SQ + s] = acc * scale;
}

__global__ __launch_bounds__(256)
void softmax_rows_kernel(float* __restrict__ x, int cols) {
  const int row = blockIdx.x;
  float* xr = x + (size_t)row * cols;
  __shared__ float red[256];
  float m = -3.4e38f;
  for (int c = threadIdx.x; c < cols; c += 256) m = fmaxf(m, xr[c]);
  red[threadIdx.x] = m;
  __syncthreads();
  for (int st = 128; st > 0; st >>= 1) {
    if (threadIdx.x < st) red[threadIdx.x] = fmaxf(red[threadIdx.x], red[threadIdx.x + st]);
    __syncthreads();
  }
  m = red[0];
  __syncthreads();
  float sum = 0.f;
  for (int c = threadIdx.x; c < cols; c += 256) {
    float e = expf(xr[c] - m);
    xr[c] = e;
    sum += e;
  }
  red[threadIdx.x] = sum;
  __syncthreads();
  for (int st = 128; st > 0; st >>= 1) {
    if (threadIdx.x < st) red[threadIdx.x] += red[threadIdx.x + st];
    __syncthreads();
  }
  float inv = 1.f / red[0];
  for (int c = threadIdx.x; c < cols; c += 256) xr[c] *= inv;
}

// ctx[b,p] = sum_s aw[b,s]*V[b,s,p]
__global__ __launch_bounds__(256)
void ctx_kernel(const float* __restrict__ aw, const _Float16* __restrict__ V16,
                float* __restrict__ ctx) {
  const int b = blockIdx.y;
  const int p = blockIdx.x * 256 + threadIdx.x;
  const float* awb = aw + (size_t)b * SQ;
  const _Float16* vb = V16 + (size_t)b * SQ * PQ + p;
  float acc = 0.f;
  for (int s = 0; s < SQ; ++s) acc += awb[s] * (float)vb[(size_t)s * PQ];
  ctx[(size_t)b * PQ + p] = acc;
}

__global__ __launch_bounds__(256)
void concat_kernel(const float* __restrict__ a, const float* __restrict__ c,
                   float* __restrict__ o) {
  const int b = blockIdx.y;
  const int p = blockIdx.x * 256 + threadIdx.x;
  o[(size_t)b * 2 * PQ + p] = a[(size_t)b * PQ + p];
  o[(size_t)b * 2 * PQ + PQ + p] = c[(size_t)b * PQ + p];
}

__global__ __launch_bounds__(256)
void add_kernel(const float* __restrict__ a, const float* __restrict__ b2,
                float* __restrict__ o, int n) {
  int i = blockIdx.x * 256 + threadIdx.x;
  if (i < n) o[i] = a[i] + b2[i];
}

// GRU cell elementwise; updates h in place, mirrors state into d_out states region
__global__ __launch_bounds__(256)
void gru_kernel(const float* __restrict__ gi, const float* __restrict__ gh,
                float* __restrict__ h, float* __restrict__ state_out) {
  const int b = blockIdx.y;
  const int p = blockIdx.x * 256 + threadIdx.x;
  const size_t base = (size_t)b * 3 * PQ;
  float ir = gi[base + p], iz = gi[base + PQ + p], inn = gi[base + 2 * PQ + p];
  float hr = gh[base + p], hz = gh[base + PQ + p], hn = gh[base + 2 * PQ + p];
  float r = 1.f / (1.f + expf(-(ir + hr)));
  float z = 1.f / (1.f + expf(-(iz + hz)));
  float n = tanhf(inn + r * hn);
  float hp = h[(size_t)b * PQ + p];
  float o = (1.f - z) * n + z * hp;
  h[(size_t)b * PQ + p] = o;
  state_out[(size_t)b * PQ + p] = o;
}

// logits[b,step] = bit_linear(plan_state, Wg[step], bg[step])  (N==1)
__global__ __launch_bounds__(256)
void logits_kernel(const int8_t* __restrict__ psq, const float* __restrict__ ps_scale,
                   const int8_t* __restrict__ wgq, const float* __restrict__ gammaSlot,
                   const float* __restrict__ bgi, float* __restrict__ logits, int step) {
  const int b = blockIdx.x;
  __shared__ int red[256];
  int s = 0;
  for (int p = threadIdx.x; p < PQ; p += 256)
    s += (int)psq[(size_t)b * PQ + p] * (int)wgq[p];
  red[threadIdx.x] = s;
  __syncthreads();
  for (int st = 128; st > 0; st >>= 1) {
    if (threadIdx.x < st) red[threadIdx.x] += red[threadIdx.x + st];
    __syncthreads();
  }
  if (threadIdx.x == 0)
    logits[(size_t)b * STEPSQ + step] =
        (float)red[0] * gammaSlot[0] * ps_scale[b] + bgi[0];
}

// flat[b, i*P+p] = states[i,b,p]
__global__ __launch_bounds__(256)
void flatten_kernel(const float* __restrict__ states, float* __restrict__ flat) {
  const int b = blockIdx.y;
  const int idx = blockIdx.x * 256 + threadIdx.x;   // [0, STEPS*P)
  const int i = idx / PQ;
  const int p = idx - i * PQ;
  flat[(size_t)b * STEPSQ * PQ + idx] = states[((size_t)i * BQ + b) * PQ + p];
}

__global__ void softmax_small_kernel(const float* __restrict__ logits,
                                     float* __restrict__ outp) {
  const int b = threadIdx.x;
  if (b >= BQ) return;
  float m = -3.4e38f;
  for (int i = 0; i < STEPSQ; ++i) m = fmaxf(m, logits[b * STEPSQ + i]);
  float e[STEPSQ];
  float sum = 0.f;
  for (int i = 0; i < STEPSQ; ++i) {
    e[i] = expf(logits[b * STEPSQ + i] - m);
    sum += e[i];
  }
  for (int i = 0; i < STEPSQ; ++i) outp[b * STEPSQ + i] = e[i] / sum;
}

// ---------------------------------------------------------------------------
// Host launch sequence
// ---------------------------------------------------------------------------

extern "C" void kernel_launch(void* const* d_in, const int* in_sizes, int n_in,
                              void* d_out, int out_size, void* d_ws, size_t ws_size,
                              hipStream_t stream) {
  (void)in_sizes; (void)n_in; (void)out_size; (void)ws_size;

  const float* context_state  = (const float*)d_in[0];
  const float* context_memory = (const float*)d_in[1];
  const float* W_enc = (const float*)d_in[2];
  const float* b_enc = (const float*)d_in[3];
  const float* W_ih  = (const float*)d_in[4];
  const float* W_hh  = (const float*)d_in[5];
  const float* b_ih  = (const float*)d_in[6];
  const float* b_hh  = (const float*)d_in[7];
  const float* Wq_w  = (const float*)d_in[8];
  const float* Wk_w  = (const float*)d_in[9];
  const float* Wv_w  = (const float*)d_in[10];
  const float* Wc_w  = (const float*)d_in[11];
  const float* bc    = (const float*)d_in[12];
  const float* W_agg = (const float*)d_in[13];
  const float* b_agg = (const float*)d_in[14];
  const float* W_out = (const float*)d_in[15];
  const float* b_out = (const float*)d_in[16];
  const float* Wg    = (const float*)d_in[17];
  const float* bg    = (const float*)d_in[18];
  float* out = (float*)d_out;

  // Bump allocator over workspace
  char* ws = (char*)d_ws;
  size_t off = 0;
  auto alloc = [&](size_t bytes) -> void* {
    off = (off + 255) & ~(size_t)255;
    void* p = ws + off;
    off += bytes;
    return p;
  };

  float* sums   = (float*)alloc(16 * sizeof(float));
  float* gammas = (float*)alloc(16 * sizeof(float));

  int8_t* Wenc_q = (int8_t*)alloc((size_t)PQ * HQ);
  int8_t* Wih_q  = (int8_t*)alloc((size_t)3 * PQ * PQ);
  int8_t* Whh_q  = (int8_t*)alloc((size_t)3 * PQ * PQ);
  int8_t* Wqq    = (int8_t*)alloc((size_t)PQ * PQ);
  int8_t* Wkq    = (int8_t*)alloc((size_t)PQ * HQ);
  int8_t* Wvq    = (int8_t*)alloc((size_t)PQ * HQ);
  int8_t* Wcq    = (int8_t*)alloc((size_t)PQ * 2 * PQ);
  int8_t* Waggq  = (int8_t*)alloc((size_t)PQ * STEPSQ * PQ);
  int8_t* Woutq  = (int8_t*)alloc((size_t)HQ * PQ);
  int8_t* Wgq    = (int8_t*)alloc((size_t)STEPSQ * PQ);

  int8_t* memq      = (int8_t*)alloc((size_t)BQ * SQ * HQ);
  float*  mem_scale = (float*)alloc((size_t)BQ * SQ * sizeof(float));
  _Float16* Kbuf    = (_Float16*)alloc((size_t)BQ * SQ * PQ * 2);
  _Float16* Vbuf    = (_Float16*)alloc((size_t)BQ * SQ * PQ * 2);

  float*  plan      = (float*)alloc((size_t)BQ * PQ * 4);
  int8_t* psq       = (int8_t*)alloc((size_t)BQ * PQ);
  float*  ps_scale  = (float*)alloc(BQ * 4);
  int8_t* csq       = (int8_t*)alloc((size_t)BQ * HQ);
  float*  cs_scale  = (float*)alloc(BQ * 4);
  float*  qbuf      = (float*)alloc((size_t)BQ * PQ * 4);
  float*  scores    = (float*)alloc((size_t)BQ * SQ * 4);
  float*  ctx       = (float*)alloc((size_t)BQ * PQ * 4);
  float*  xcat      = (float*)alloc((size_t)BQ * 2 * PQ * 4);
  int8_t* xcatq     = (int8_t*)alloc((size_t)BQ * 2 * PQ);
  float*  xcat_scale= (float*)alloc(BQ * 4);
  float*  comb      = (float*)alloc((size_t)BQ * PQ * 4);
  float*  xbuf      = (float*)alloc((size_t)BQ * PQ * 4);
  int8_t* xq2       = (int8_t*)alloc((size_t)BQ * PQ);
  float*  x2_scale  = (float*)alloc(BQ * 4);
  float*  gi        = (float*)alloc((size_t)BQ * 3 * PQ * 4);
  float*  gh        = (float*)alloc((size_t)BQ * 3 * PQ * 4);
  float*  logits    = (float*)alloc((size_t)BQ * STEPSQ * 4);
  float*  flat      = (float*)alloc((size_t)BQ * STEPSQ * PQ * 4);
  int8_t* flatq     = (int8_t*)alloc((size_t)BQ * STEPSQ * PQ);
  float*  flat_scale= (float*)alloc(BQ * 4);
  float*  aggr      = (float*)alloc((size_t)BQ * PQ * 4);
  int8_t* aggq      = (int8_t*)alloc((size_t)BQ * PQ);
  float*  agg_scale = (float*)alloc(BQ * 4);

  float* states_out = out + (size_t)BQ * HQ;                       // [STEPS,B,P]
  float* imp_out    = states_out + (size_t)STEPSQ * BQ * PQ;       // [B,STEPS]

  // ---- Pass A: weight quantization --------------------------------------
  init_slots_kernel<<<1, 32, 0, stream>>>(sums);

  struct WQEnt { const float* w; long long n; int slot; int8_t* dst; };
  WQEnt wlist[14] = {
      {W_enc, (long long)PQ * HQ, 0, Wenc_q},
      {W_ih, 3LL * PQ * PQ, 1, Wih_q},
      {W_hh, 3LL * PQ * PQ, 2, Whh_q},
      {Wq_w, (long long)PQ * PQ, 3, Wqq},
      {Wk_w, (long long)PQ * HQ, 4, Wkq},
      {Wv_w, (long long)PQ * HQ, 5, Wvq},
      {Wc_w, 2LL * PQ * PQ, 6, Wcq},
      {W_agg, (long long)STEPSQ * PQ * PQ, 7, Waggq},
      {W_out, (long long)HQ * PQ, 8, Woutq},
      {Wg + 0 * PQ, PQ, 9, Wgq + 0 * PQ},
      {Wg + 1 * PQ, PQ, 10, Wgq + 1 * PQ},
      {Wg + 2 * PQ, PQ, 11, Wgq + 2 * PQ},
      {Wg + 3 * PQ, PQ, 12, Wgq + 3 * PQ},
      {Wg + 4 * PQ, PQ, 13, Wgq + 4 * PQ},
  };
  for (int i = 0; i < 14; ++i) {
    long long nb = (wlist[i].n + 255) / 256;
    if (nb > 2048) nb = 2048;
    abssum_kernel<<<(int)nb, 256, 0, stream>>>(wlist[i].w, wlist[i].n, sums + wlist[i].slot);
  }
  for (int i = 0; i < 14; ++i) {
    long long nb = (wlist[i].n + 255) / 256;
    quantize_weight_kernel<<<(int)nb, 256, 0, stream>>>(
        wlist[i].w, wlist[i].n, sums + wlist[i].slot, gammas + wlist[i].slot, wlist[i].dst);
  }

  auto launch_gemm = [&](const int8_t* Xq, const float* xs, const int8_t* Wqp,
                         const float* gslot, const float* bias, int M, int N, int K,
                         int act, float* o32, _Float16* o16) {
    dim3 g(N / GT_N, (M + GT_M - 1) / GT_M);
    gemm_i8_wmma_kernel<<<g, 256, 0, stream>>>(Xq, xs, Wqp, gslot, bias,
                                               M, N, K, act, o32, o16);
  };

  // ---- Pass B: K/V projections (dominant GEMMs, int8 WMMA, fp16 out) ----
  quantize_rows_kernel<<<BQ * SQ, 256, 0, stream>>>(context_memory, HQ, memq, mem_scale);
  launch_gemm(memq, mem_scale, Wkq, gammas + 4, nullptr, BQ * SQ, PQ, HQ, 0, nullptr, Kbuf);
  launch_gemm(memq, mem_scale, Wvq, gammas + 5, nullptr, BQ * SQ, PQ, HQ, 0, nullptr, Vbuf);

  // ---- Encoder: plan_state = tanh(bit_linear(context_state, W_enc, b_enc))
  quantize_rows_kernel<<<BQ, 256, 0, stream>>>(context_state, HQ, csq, cs_scale);
  launch_gemm(csq, cs_scale, Wenc_q, gammas + 0, b_enc, BQ, PQ, HQ, 1, plan, nullptr);
  quantize_rows_kernel<<<BQ, 256, 0, stream>>>(plan, PQ, psq, ps_scale);

  const float scale = 1.0f / sqrtf((float)PQ);

  // ---- Planning steps ----------------------------------------------------
  for (int i = 0; i < STEPSQ; ++i) {
    // q = bit_linear(plan_state, Wq)
    launch_gemm(psq, ps_scale, Wqq, gammas + 3, nullptr, BQ, PQ, PQ, 0, qbuf, nullptr);
    // attention
    scores_kernel<<<dim3(SQ / 8, BQ), 256, 0, stream>>>(qbuf, Kbuf, scores, scale);
    softmax_rows_kernel<<<BQ, 256, 0, stream>>>(scores, SQ);
    ctx_kernel<<<dim3(PQ / 256, BQ), 256, 0, stream>>>(scores, Vbuf, ctx);
    // comb = bit_linear([plan,ctx], Wc, bc); x = plan + comb
    concat_kernel<<<dim3(PQ / 256, BQ), 256, 0, stream>>>(plan, ctx, xcat);
    quantize_rows_kernel<<<BQ, 256, 0, stream>>>(xcat, 2 * PQ, xcatq, xcat_scale);
    launch_gemm(xcatq, xcat_scale, Wcq, gammas + 6, bc, BQ, PQ, 2 * PQ, 0, comb, nullptr);
    add_kernel<<<(BQ * PQ) / 256, 256, 0, stream>>>(plan, comb, xbuf, BQ * PQ);
    // GRU gates
    quantize_rows_kernel<<<BQ, 256, 0, stream>>>(xbuf, PQ, xq2, x2_scale);
    launch_gemm(xq2, x2_scale, Wih_q, gammas + 1, b_ih, BQ, 3 * PQ, PQ, 0, gi, nullptr);
    launch_gemm(psq, ps_scale, Whh_q, gammas + 2, b_hh, BQ, 3 * PQ, PQ, 0, gh, nullptr);
    gru_kernel<<<dim3(PQ / 256, BQ), 256, 0, stream>>>(
        gi, gh, plan, states_out + (size_t)i * BQ * PQ);
    // re-quantize updated plan_state (used by logits + next step)
    quantize_rows_kernel<<<BQ, 256, 0, stream>>>(plan, PQ, psq, ps_scale);
    logits_kernel<<<BQ, 256, 0, stream>>>(psq, ps_scale, Wgq + (size_t)i * PQ,
                                          gammas + 9 + i, bg + i, logits, i);
  }

  // ---- Aggregation + output ---------------------------------------------
  flatten_kernel<<<dim3((STEPSQ * PQ) / 256, BQ), 256, 0, stream>>>(states_out, flat);
  quantize_rows_kernel<<<BQ, 256, 0, stream>>>(flat, STEPSQ * PQ, flatq, flat_scale);
  launch_gemm(flatq, flat_scale, Waggq, gammas + 7, b_agg, BQ, PQ, STEPSQ * PQ, 1,
              aggr, nullptr);
  quantize_rows_kernel<<<BQ, 256, 0, stream>>>(aggr, PQ, aggq, agg_scale);
  launch_gemm(aggq, agg_scale, Woutq, gammas + 8, b_out, BQ, HQ, PQ, 0, out, nullptr);

  softmax_small_kernel<<<1, 32, 0, stream>>>(logits, imp_out);
}
